// WindowAttention_56195352101036
// MI455X (gfx1250) — matrix operand order repaired and checked by reading, not compiled, not run on hardware
//
#include <hip/hip_runtime.h>
#include <hip/hip_bf16.h>

typedef _Float16 half_t;
typedef __attribute__((ext_vector_type(16))) _Float16 v16h;
typedef __attribute__((ext_vector_type(8)))  _Float16 v8h;
typedef __attribute__((ext_vector_type(8)))  float    v8f;

#define NTOK 49
#define DIMC 384
#define NH   12
#define HD   32
#define MP   64
#define TD   1152
#define QKV_PITCH 1168   // halves
#define XO_PITCH  400    // halves
#define SS_PITCH  68     // floats
#define SP_PITCH  72     // halves

#define OFF_QKV  0
#define OFF_XO   149504            // 64*1168*2
#define OFF_SS   200704            // +64*400*2
#define OFF_SP   235520            // +8*16*68*4
#define SMEM_BYTES 253952          // +8*16*72*2

#define WMMA_F16(a, b, c) \
  __builtin_amdgcn_wmma_f32_16x16x32_f16(false, (a), false, (b), (short)0, (c), false, false)

// A fragment (16x32 f16, row-major source): lane m<16 holds K 0..15,
// lane m+16 holds K 16..31  -> two contiguous 16B chunks per lane.
__device__ __forceinline__ v16h load_a_frag(const half_t* base, int pitch, int row0, int col0) {
  const int lane = threadIdx.x & 31;
  const half_t* p = base + (row0 + (lane & 15)) * pitch + col0 + ((lane >> 4) << 4);
  v8h lo = *(const v8h*)p;
  v8h hi = *(const v8h*)(p + 8);
  return __builtin_shufflevector(lo, hi, 0,1,2,3,4,5,6,7,8,9,10,11,12,13,14,15);
}

// B fragment (32x16 f16): lane n<16 = column n K 0..15, lane n+16 = column n K 16..31.
// Source indexed [col][k] row-major (i.e. W stored as (out, in)) -> one 32B load.
__device__ __forceinline__ v16h load_b_frag(const half_t* base, int pitch, int col0, int k0) {
  const int lane = threadIdx.x & 31;
  return *(const v16h*)(base + (col0 + (lane & 15)) * pitch + k0 + ((lane >> 4) << 4));
}

// B fragment when source is indexed [k][col] (strided gather, used for V).
__device__ __forceinline__ v16h load_b_frag_strided(const half_t* base, int pitch, int col0, int k0) {
  const int lane = threadIdx.x & 31;
  const int c = col0 + (lane & 15);
  const int k = k0 + ((lane >> 4) << 4);
  v16h r;
#pragma unroll
  for (int e = 0; e < 16; ++e) r[e] = base[(k + e) * pitch + c];
  return r;
}

__global__ void prep_kernel(const float* __restrict__ qkv_w, const float* __restrict__ proj_w,
                            const float* __restrict__ rpb_table,
                            half_t* __restrict__ wqkv, half_t* __restrict__ wproj,
                            float* __restrict__ rpb) {
  const int i = blockIdx.x * 256 + threadIdx.x;
  if (i < TD * DIMC)   wqkv[i]  = (half_t)qkv_w[i];
  if (i < DIMC * DIMC) wproj[i] = (half_t)proj_w[i];
  if (i < NH * NTOK * NTOK) {
    int h = i / (NTOK * NTOK);
    int rem = i - h * NTOK * NTOK;
    int a  = rem / NTOK;
    int bb = rem - a * NTOK;
    int r1 = a / 7,  c1 = a  - r1 * 7;
    int r2 = bb / 7, c2 = bb - r2 * 7;
    int rel = (r1 - r2 + 6) * 13 + (c1 - c2 + 6);
    rpb[i] = rpb_table[rel * NH + h];
  }
}

__global__ void latency_kernel(const float* __restrict__ gw, const int* __restrict__ H,
                               const int* __restrict__ W, float* __restrict__ out_lat) {
  if (threadIdx.x == 0 && blockIdx.x == 0) {
    float nW = (float)(H[0] * W[0]) / 49.0f;
    const int alpha[4] = {1, 2, 4, 8};
    float lat = 0.0f;
    for (int i = 0; i < 4; ++i) {
      float a = (float)alpha[i];
      float flops = 4.0f * 49.0f * (384.0f / a) * (384.0f / a)
                  + 2.0f * (12.0f / a) * 49.0f * 49.0f * 32.0f;
      lat += flops * floorf(nW / a) * gw[i];
    }
    *out_lat = lat;
  }
}

__global__ __launch_bounds__(256)
void fused_window_attn(const float* __restrict__ x, const float* __restrict__ gw,
                       const float* __restrict__ qkv_b, const float* __restrict__ proj_b,
                       const half_t* __restrict__ wqkv, const half_t* __restrict__ wproj,
                       const float* __restrict__ rpb, float* __restrict__ out) {
  extern __shared__ char smem[];
  half_t* qkv = (half_t*)(smem + OFF_QKV);   // [64][1168] : q | k | v
  half_t* xo  = (half_t*)(smem + OFF_XO);    // [64][400]  : x (f16) then O
  float*  sS  = (float*)(smem + OFF_SS);     // [8][16][68] per-wave scores
  half_t* sP  = (half_t*)(smem + OFF_SP);    // [8][16][72] per-wave probs

  const int b    = blockIdx.x;
  const int tid  = threadIdx.x;
  const int wave = tid >> 5;
  const int lane = tid & 31;
  const int lhi  = lane >> 4;
  const int lm   = lane & 15;

  // ---- Phase 1: stage x -> f16 LDS, zero pad rows 49..63 ----
  const float* xb = x + (size_t)b * NTOK * DIMC;
  for (int i = tid * 4; i < MP * DIMC; i += 1024) {
    int r = i / DIMC, c = i - r * DIMC;
    half_t* dst = xo + r * XO_PITCH + c;
    if (r < NTOK) {
      const float4 v = *(const float4*)(xb + r * DIMC + c);
      dst[0] = (half_t)v.x; dst[1] = (half_t)v.y;
      dst[2] = (half_t)v.z; dst[3] = (half_t)v.w;
    } else {
      dst[0] = (half_t)0.0f; dst[1] = (half_t)0.0f;
      dst[2] = (half_t)0.0f; dst[3] = (half_t)0.0f;
    }
  }
  __syncthreads();

  // ---- Phase 2: QKV = x @ Wqkv^T (+bias, q pre-scaled), rows>=49 zeroed ----
  const float scale = 0.17677669529663687f;  // 32^-0.5
  for (int mt = 0; mt < 4; ++mt) {
    v16h afr[12];
#pragma unroll
    for (int kt = 0; kt < 12; ++kt) afr[kt] = load_a_frag(xo, XO_PITCH, mt * 16, kt * 32);
    for (int nt = wave; nt < 72; nt += 8) {
      v8f acc = {};
#pragma unroll
      for (int kt = 0; kt < 12; ++kt) {
        v16h bf = load_b_frag(wqkv, DIMC, nt * 16, kt * 32);
        acc = WMMA_F16(afr[kt], bf, acc);
      }
      const int o = nt * 16 + lm;
      const float bias = qkv_b[o];
      const float mul  = (o < DIMC) ? scale : 1.0f;
#pragma unroll
      for (int j = 0; j < 8; ++j) {
        int r = mt * 16 + j + 8 * lhi;
        float v = (acc[j] + bias) * mul;
        qkv[r * QKV_PITCH + o] = (half_t)((r < NTOK) ? v : 0.0f);
      }
    }
  }
  __syncthreads();

  // ---- Phase 3: attention, 48 units = (head, row-tile), 6 per wave ----
  for (int u = 0; u < 6; ++u) {
    const int unit = wave + u * 8;
    const int h = unit >> 2;
    const int mt = unit & 3;
    float* sSw = sS + wave * 16 * SS_PITCH;

    v16h qa = load_a_frag(qkv, QKV_PITCH, mt * 16, h * HD);
#pragma unroll
    for (int ct = 0; ct < 4; ++ct) {
      v8f z = {};
      v16h kb = load_b_frag(qkv, QKV_PITCH, ct * 16, DIMC + h * HD);
      v8f s = WMMA_F16(qa, kb, z);
#pragma unroll
      for (int j = 0; j < 8; ++j) {
        int i  = mt * 16 + j + 8 * lhi;
        int jj = ct * 16 + lm;
        float v = (i < NTOK && jj < NTOK)
                    ? s[j] + rpb[h * NTOK * NTOK + i * NTOK + jj]
                    : -1.0e30f;
        sSw[(j + 8 * lhi) * SS_PITCH + ct * 16 + lm] = v;
      }
    }
    __syncthreads();

    {  // softmax: lane -> row lm, column half lhi
      float* rowp = sSw + lm * SS_PITCH + lhi * 32;
      float mx = -3.0e38f;
#pragma unroll
      for (int c = 0; c < 32; ++c) mx = fmaxf(mx, rowp[c]);
      mx = fmaxf(mx, __shfl_xor(mx, 16, 32));
      float ev[32], sum = 0.0f;
#pragma unroll
      for (int c = 0; c < 32; ++c) { float e = __expf(rowp[c] - mx); ev[c] = e; sum += e; }
      sum += __shfl_xor(sum, 16, 32);
      float inv = 1.0f / sum;
      half_t* pp = sP + wave * 16 * SP_PITCH + lm * SP_PITCH + lhi * 32;
#pragma unroll
      for (int c = 0; c < 32; ++c) pp[c] = (half_t)(ev[c] * inv);
    }
    __syncthreads();

#pragma unroll
    for (int dt = 0; dt < 2; ++dt) {  // O = P @ V, K over 64 tokens
      v8f acc = {};
#pragma unroll
      for (int kt = 0; kt < 2; ++kt) {
        v16h pa = load_a_frag(sP + wave * 16 * SP_PITCH, SP_PITCH, 0, kt * 32);
        v16h vb = load_b_frag_strided(qkv, QKV_PITCH, 2 * DIMC + h * HD + dt * 16, kt * 32);
        acc = WMMA_F16(pa, vb, acc);
      }
#pragma unroll
      for (int j = 0; j < 8; ++j) {
        int r = mt * 16 + j + 8 * lhi;
        xo[r * XO_PITCH + h * HD + dt * 16 + lm] = (half_t)acc[j];
      }
    }
    __syncthreads();
  }

  // ---- Phase 4: projection with K-split checkpoints for branch prefixes ----
  const float g0 = gw[0], g1 = gw[1], g2 = gw[2], g3 = gw[3];
  for (int nti = 0; nti < 3; ++nti) {
    const int nt = wave + nti * 8;
    for (int mt = 0; mt < 4; ++mt) {
      v8f acc = {}, p1 = {}, p2 = {}, p3 = {};
#pragma unroll
      for (int kt = 0; kt < 12; ++kt) {
        v16h oa = load_a_frag(xo, XO_PITCH, mt * 16, kt * 32);
        v16h wb = load_b_frag(wproj, DIMC, nt * 16, kt * 32);
        acc = WMMA_F16(oa, wb, acc);
        if (kt == 0) p1 = acc;   // d < 32   (alpha=8)
        if (kt == 2) p2 = acc;   // d < 96   (alpha=4)
        if (kt == 5) p3 = acc;   // d < 192  (alpha=2)
      }
      const int c = nt * 16 + lm;
      const float bias = proj_b[c];
      const float m48  = (c < 48)  ? g3 : 0.0f;
      const float m96  = (c < 96)  ? g2 : 0.0f;
      const float m192 = (c < 192) ? g1 : 0.0f;
#pragma unroll
      for (int j = 0; j < 8; ++j) {
        int r = mt * 16 + j + 8 * lhi;
        if (r < NTOK) {
          float val;
          if (b < 512) {
            val = acc[j] + bias;
          } else {
            val = m48  * (p1[j] + bias) + m96 * (p2[j] + bias)
                + m192 * (p3[j] + bias) + g0  * (acc[j] + bias);
          }
          out[(size_t)b * NTOK * DIMC + r * DIMC + c] = val;
        }
      }
    }
  }
}

extern "C" void kernel_launch(void* const* d_in, const int* in_sizes, int n_in,
                              void* d_out, int out_size, void* d_ws, size_t ws_size,
                              hipStream_t stream) {
  (void)in_sizes; (void)n_in; (void)ws_size;
  const float* x         = (const float*)d_in[0];
  const float* gw        = (const float*)d_in[1];
  const float* qkv_w     = (const float*)d_in[2];
  const float* qkv_b     = (const float*)d_in[3];
  const float* proj_w    = (const float*)d_in[4];
  const float* proj_b    = (const float*)d_in[5];
  const float* rpb_table = (const float*)d_in[6];
  const int*   Hp        = (const int*)d_in[7];
  const int*   Wp        = (const int*)d_in[8];
  float* out = (float*)d_out;

  // workspace layout
  char* ws = (char*)d_ws;
  half_t* wqkv  = (half_t*)(ws);                       // 1152*384 halves
  half_t* wproj = (half_t*)(ws + 884736);              // 384*384 halves
  float*  rpb   = (float*)(ws + 884736 + 294912);      // 12*49*49 floats

  hipFuncSetAttribute((const void*)fused_window_attn,
                      hipFuncAttributeMaxDynamicSharedMemorySize, SMEM_BYTES);

  prep_kernel<<<(TD * DIMC + 255) / 256, 256, 0, stream>>>(
      qkv_w, proj_w, rpb_table, wqkv, wproj, rpb);
  fused_window_attn<<<1024, 256, SMEM_BYTES, stream>>>(
      x, gw, qkv_b, proj_b, wqkv, wproj, rpb, out);
  latency_kernel<<<1, 1, 0, stream>>>(gw, Hp, Wp, out + (out_size - 1));
}
